// RSSM_20753281974655
// MI455X (gfx1250) — compile-verified, matrix operand order checked by reading, not run on hardware
//
#include <hip/hip_runtime.h>
#include <hip/hip_bf16.h>
#include <math.h>

// ---------------------------------------------------------------------------
// RSSM scan (Dreamer) for MI455X / gfx1250.
//  - wave32 WMMA bf16 (f32 accumulate), weights converted once to bf16
//    (~21 MB -> L2 resident across all 50 steps)
//  - activation panels staged to LDS with global_load_async_to_lds_b128
//    (ASYNCcnt) and shared by all waves of a block; register blocking NB=4
//  - 50 sequential steps of 7 fused kernels on one stream
// ---------------------------------------------------------------------------

#define T_STEPS 50
#define BATCH   512
#define EMB     1536
#define ACTD    64
#define DET     1024
#define STO     128
#define SA_DIM  (STO + ACTD)   // 192
#define QK      (DET + EMB)    // 2560
#define MIN_STD 0.1f

typedef __bf16 bf16_t;
typedef __attribute__((ext_vector_type(16))) __bf16 v16bf;
typedef __attribute__((ext_vector_type(8)))  float  v8f;

// ---- WMMA fragment helpers (CDNA5 16-bit A/B layout, f32 C/D layout) -------

__device__ __forceinline__ int frag_k(int i, int lane) {
  // halfword i (0..15) -> K index for 16-bit A (16x32) / B (32x16) operands
  return ((i >= 8) ? 16 : 0) + ((lane & 16) ? 8 : 0) + (((i >> 1) & 3) << 1) + (i & 1);
}

__device__ __forceinline__ v16bf load_frag_bf16(const bf16_t* __restrict__ base,
                                                int ld, int r0, int k0, int lane) {
  const bf16_t* p = base + (size_t)(r0 + (lane & 15)) * ld + k0;
  v16bf f;
#pragma unroll
  for (int i = 0; i < 16; ++i) f[i] = p[frag_k(i, lane)];
  return f;
}

// A fragment from an LDS-staged panel [16 rows x ld elems]
__device__ __forceinline__ v16bf load_frag_lds(const bf16_t* sA, int ld, int k0, int lane) {
  const bf16_t* p = sA + (lane & 15) * ld + k0;
  v16bf f;
#pragma unroll
  for (int i = 0; i < 16; ++i) f[i] = p[frag_k(i, lane)];
  return f;
}

__device__ __forceinline__ v16bf load_frag_f32(const float* __restrict__ base,
                                               int ld, int r0, int k0, int lane) {
  // on-the-fly fp32 -> bf16 conversion (obs_embed is read exactly once)
  const float* p = base + (size_t)(r0 + (lane & 15)) * ld + k0;
  v16bf f;
#pragma unroll
  for (int i = 0; i < 16; ++i) f[i] = (bf16_t)p[frag_k(i, lane)];
  return f;
}

__device__ __forceinline__ v8f wmma_bf(v16bf a, v16bf b, v8f c) {
  return __builtin_amdgcn_wmma_f32_16x16x32_bf16(false, a, false, b, (short)0, c,
                                                 false, false);
}

// ---- async global -> LDS staging (CDNA5 VGLOBAL async ops, ASYNCcnt) -------

__device__ __forceinline__ void async_copy16(unsigned lds_off, const void* gaddr) {
  asm volatile("global_load_async_to_lds_b128 %0, %1, off"
               :: "v"(lds_off), "v"(gaddr) : "memory");
}

// Stage `rows` x `ld` bf16 elements (row r taken from gsrc row r0+r) into sdst.
// All threads of the block participate; 16B per lane per issue.
__device__ __forceinline__ void stage_panel_async(bf16_t* sdst,
                                                  const bf16_t* __restrict__ gsrc,
                                                  int rows, int ld, int r0) {
  const int tid = threadIdx.x, nthr = blockDim.x;
  const int chunks_per_row = (ld * 2) >> 4;           // 16B chunks per row
  const int total = rows * chunks_per_row;
  for (int c = tid; c < total; c += nthr) {
    int r = c / chunks_per_row, x = c - r * chunks_per_row;
    unsigned lds_off = (unsigned)(uintptr_t)(sdst + r * ld) + (unsigned)(x << 4);
    const char* g = (const char*)(gsrc + (size_t)(r0 + r) * ld) + (x << 4);
    async_copy16(lds_off, g);
  }
  asm volatile("s_wait_asynccnt 0" ::: "memory");
  __syncthreads();
}

// ---- scalar math ------------------------------------------------------------

__device__ __forceinline__ float elu_f(float x)      { return x > 0.0f ? x : (__expf(x) - 1.0f); }
__device__ __forceinline__ float sigmoid_f(float x)  { return 1.0f / (1.0f + __expf(-x)); }
__device__ __forceinline__ float softplus_f(float x) {
  return fmaxf(x, 0.0f) + log1pf(__expf(-fabsf(x)));
}

// ---- small utility kernels --------------------------------------------------

__global__ void cvt_f32_bf16(const float* __restrict__ src, bf16_t* __restrict__ dst, int n) {
  int i = blockIdx.x * blockDim.x + threadIdx.x;
  if (i < n) dst[i] = (bf16_t)src[i];
}

__global__ void init_state(const float* __restrict__ prev_stoch,
                           const float* __restrict__ prev_deter,
                           float* __restrict__ stoch_state,
                           float* __restrict__ deter_state) {
  int i = blockIdx.x * blockDim.x + threadIdx.x;
  if (i < BATCH * STO) stoch_state[i] = prev_stoch[i];
  else {
    int j = i - BATCH * STO;
    if (j < BATCH * DET) deter_state[j] = prev_deter[j];
  }
}

// Build per-step inputs: sa = [stoch*nt, act*nt] (bf16), h = deter*nt (bf16+f32)
__global__ void prep_step(const float* __restrict__ stoch_state,
                          const float* __restrict__ action_t,
                          const float* __restrict__ nt_t,
                          const float* __restrict__ deter_state,
                          bf16_t* __restrict__ sa_bf,
                          bf16_t* __restrict__ h_bf,
                          float*  __restrict__ h_f32) {
  int i = blockIdx.x * blockDim.x + threadIdx.x;
  const int total_sa = BATCH * SA_DIM;
  if (i < total_sa) {
    int b = i / SA_DIM, c = i % SA_DIM;
    float nt = nt_t[b];
    float v = (c < STO) ? stoch_state[b * STO + c] : action_t[b * ACTD + (c - STO)];
    sa_bf[i] = (bf16_t)(v * nt);
  } else {
    int j = i - total_sa;
    if (j < BATCH * DET) {
      int b = j / DET;
      float v = deter_state[j] * nt_t[b];
      h_bf[j]  = (bf16_t)v;
      h_f32[j] = v;
    }
  }
}

// ---- Y = elu(X @ W^T + bias), bf16 in / bf16 out ---------------------------
// N == 1024. Block = 8 waves, one 16-row M tile (LDS-staged A panel).
// Wave w covers 4 N tiles: n0 = nb*512 + w*64 .. +3*16.  Grid = 32 * (N/512).
__global__ void gemm_bias_elu(const bf16_t* __restrict__ X,
                              const bf16_t* __restrict__ W,
                              const float*  __restrict__ bias,
                              bf16_t* __restrict__ Y, int N, int K) {
  extern __shared__ __align__(16) char smem_raw[];
  bf16_t* sA = (bf16_t*)smem_raw;                        // [16][K]

  const int nb_n = N >> 9;                               // N / 512
  const int mb = blockIdx.x / nb_n, nb = blockIdx.x % nb_n;
  const int m0 = mb << 4;
  const int lane = threadIdx.x & 31, w = threadIdx.x >> 5;

  stage_panel_async(sA, X, 16, K, m0);

  const int n0 = (nb << 9) + (w << 6);                   // wave covers n0..n0+63
  v8f acc[4] = {v8f{}, v8f{}, v8f{}, v8f{}};
  for (int k0 = 0; k0 < K; k0 += 32) {
    v16bf a = load_frag_lds(sA, K, k0, lane);
#pragma unroll
    for (int j = 0; j < 4; ++j) {
      v16bf b = load_frag_bf16(W, K, n0 + (j << 4), k0, lane);
      acc[j] = wmma_bf(a, b, acc[j]);
    }
  }

  const int n = lane & 15, mrb = (lane & 16) ? 8 : 0;
#pragma unroll
  for (int j = 0; j < 4; ++j) {
    const int ng = n0 + (j << 4) + n;
    const float bv = bias[ng];
#pragma unroll
    for (int v = 0; v < 8; ++v) {
      float y = elu_f(acc[j][v] + bv);
      Y[(size_t)(m0 + mrb + v) * N + ng] = (bf16_t)y;
    }
  }
}

// ---- qh = elu([deter, obs] @ Wq1^T + bq1) ----------------------------------
// deter panel LDS-staged; obs part converted fp32->bf16 in-register.
__global__ void gemm_posterior_hidden(const bf16_t* __restrict__ deter_bf,
                                      const float*  __restrict__ obs_t,
                                      const bf16_t* __restrict__ Wq1,
                                      const float*  __restrict__ bq1,
                                      bf16_t* __restrict__ qh_bf) {
  extern __shared__ __align__(16) char smem_raw[];
  bf16_t* sA = (bf16_t*)smem_raw;                        // [16][DET]

  const int nb_n = DET >> 9;                             // 2
  const int mb = blockIdx.x / nb_n, nb = blockIdx.x % nb_n;
  const int m0 = mb << 4;
  const int lane = threadIdx.x & 31, w = threadIdx.x >> 5;

  stage_panel_async(sA, deter_bf, 16, DET, m0);

  const int n0 = (nb << 9) + (w << 6);
  v8f acc[4] = {v8f{}, v8f{}, v8f{}, v8f{}};
  for (int k0 = 0; k0 < QK; k0 += 32) {
    v16bf a = (k0 < DET) ? load_frag_lds(sA, DET, k0, lane)
                         : load_frag_f32(obs_t, EMB, m0, k0 - DET, lane);
#pragma unroll
    for (int j = 0; j < 4; ++j) {
      v16bf b = load_frag_bf16(Wq1, QK, n0 + (j << 4), k0, lane);
      acc[j] = wmma_bf(a, b, acc[j]);
    }
  }

  const int n = lane & 15, mrb = (lane & 16) ? 8 : 0;
#pragma unroll
  for (int j = 0; j < 4; ++j) {
    const int ng = n0 + (j << 4) + n;
    const float bv = bq1[ng];
#pragma unroll
    for (int v = 0; v < 8; ++v) {
      float y = elu_f(acc[j][v] + bv);
      qh_bf[(size_t)(m0 + mrb + v) * DET + ng] = (bf16_t)y;
    }
  }
}

// ---- fused GRU cell: 6 GEMM accumulators in one K-loop ---------------------
// Block = (m tile, group of 8 n tiles); x/h panels staged in LDS (2x32KB).
__global__ void gru_step(const bf16_t* __restrict__ Xbf,   // [B,D] elu(embed)
                         const bf16_t* __restrict__ Hbf,   // [B,D] deter*nt bf16
                         const float*  __restrict__ Hf32,  // [B,D] deter*nt f32
                         const bf16_t* __restrict__ Wih,   // [3D,D]
                         const bf16_t* __restrict__ Whh,   // [3D,D]
                         const float*  __restrict__ bih,
                         const float*  __restrict__ bhh,
                         float*  __restrict__ deter_state, // [B,D]
                         bf16_t* __restrict__ deter_bf,    // [B,D]
                         float*  __restrict__ out_prior_deter,
                         float*  __restrict__ out_post_deter) {
  extern __shared__ __align__(16) char smem_raw[];
  bf16_t* sX = (bf16_t*)smem_raw;                        // [16][DET]
  bf16_t* sH = sX + 16 * DET;                            // [16][DET]

  const int mb = blockIdx.x >> 3, ngrp = blockIdx.x & 7;
  const int m0 = mb << 4;
  const int lane = threadIdx.x & 31, w = threadIdx.x >> 5;

  stage_panel_async(sX, Xbf, 16, DET, m0);
  stage_panel_async(sH, Hbf, 16, DET, m0);

  const int n0 = ((ngrp << 3) + w) << 4;                 // this wave's n tile

  v8f aXR = {}, aXZ = {}, aXN = {}, aHR = {}, aHZ = {}, aHN = {};
  for (int k0 = 0; k0 < DET; k0 += 32) {
    v16bf ax = load_frag_lds(sX, DET, k0, lane);
    v16bf ah = load_frag_lds(sH, DET, k0, lane);
    v16bf br = load_frag_bf16(Wih, DET, n0,           k0, lane);
    v16bf bz = load_frag_bf16(Wih, DET, DET + n0,     k0, lane);
    v16bf bn = load_frag_bf16(Wih, DET, 2 * DET + n0, k0, lane);
    aXR = wmma_bf(ax, br, aXR);
    aXZ = wmma_bf(ax, bz, aXZ);
    aXN = wmma_bf(ax, bn, aXN);
    v16bf cr = load_frag_bf16(Whh, DET, n0,           k0, lane);
    v16bf cz = load_frag_bf16(Whh, DET, DET + n0,     k0, lane);
    v16bf cn = load_frag_bf16(Whh, DET, 2 * DET + n0, k0, lane);
    aHR = wmma_bf(ah, cr, aHR);
    aHZ = wmma_bf(ah, cz, aHZ);
    aHN = wmma_bf(ah, cn, aHN);
  }

  const int n = lane & 15, mrb = (lane & 16) ? 8 : 0;
  const int ng = n0 + n;
  const float bir = bih[ng],          bhr = bhh[ng];
  const float biz = bih[DET + ng],    bhz = bhh[DET + ng];
  const float bin = bih[2*DET + ng],  bhn = bhh[2*DET + ng];
#pragma unroll
  for (int v = 0; v < 8; ++v) {
    const int m = m0 + mrb + v;
    float r  = sigmoid_f(aXR[v] + bir + aHR[v] + bhr);
    float z  = sigmoid_f(aXZ[v] + biz + aHZ[v] + bhz);
    float nn = tanhf((aXN[v] + bin) + r * (aHN[v] + bhn));
    float h  = Hf32[(size_t)m * DET + ng];
    float d  = (1.0f - z) * nn + z * h;
    size_t idx = (size_t)m * DET + ng;
    deter_state[idx]     = d;
    deter_bf[idx]        = (bf16_t)d;
    out_prior_deter[idx] = d;
    out_post_deter[idx]  = d;
  }
}

// ---- head: p = H @ W2^T + b2 ; mean/std/reparam-sample in one pass ---------
// Block = one m tile (H panel staged in LDS); wave w -> n tile w (STO/16 = 8).
__global__ void head_step(const bf16_t* __restrict__ H,     // [B,DET]
                          const bf16_t* __restrict__ W2,    // [2*STO,DET]
                          const float*  __restrict__ b2,
                          const float*  __restrict__ eps,   // [B,STO]
                          float* __restrict__ out_mean,
                          float* __restrict__ out_std,
                          float* __restrict__ out_stoch,
                          float* __restrict__ stoch_state) { // nullable
  extern __shared__ __align__(16) char smem_raw[];
  bf16_t* sA = (bf16_t*)smem_raw;                        // [16][DET]

  const int m0 = blockIdx.x << 4;
  const int lane = threadIdx.x & 31, w = threadIdx.x >> 5;

  stage_panel_async(sA, H, 16, DET, m0);

  const int n0 = w << 4;
  v8f accM = {}, accR = {};
  for (int k0 = 0; k0 < DET; k0 += 32) {
    v16bf a  = load_frag_lds(sA, DET, k0, lane);
    v16bf bm = load_frag_bf16(W2, DET, n0,        k0, lane);
    v16bf br = load_frag_bf16(W2, DET, STO + n0,  k0, lane);
    accM = wmma_bf(a, bm, accM);
    accR = wmma_bf(a, br, accR);
  }

  const int n = lane & 15, mrb = (lane & 16) ? 8 : 0;
  const float bm = b2[n0 + n], brw = b2[STO + n0 + n];
#pragma unroll
  for (int v = 0; v < 8; ++v) {
    const int m = m0 + mrb + v;
    const size_t idx = (size_t)m * STO + n0 + n;
    float mean = accM[v] + bm;
    float sd   = softplus_f(accR[v] + brw) + MIN_STD;
    float st   = mean + sd * eps[idx];
    out_mean[idx]  = mean;
    out_std[idx]   = sd;
    out_stoch[idx] = st;
    if (stoch_state) stoch_state[idx] = st;
  }
}

// ---------------------------------------------------------------------------

extern "C" void kernel_launch(void* const* d_in, const int* in_sizes, int n_in,
                              void* d_out, int out_size, void* d_ws, size_t ws_size,
                              hipStream_t stream) {
  const float* obs    = (const float*)d_in[0];   // [T,B,E]
  const float* action = (const float*)d_in[1];   // [T,B,A]
  const float* nonterm= (const float*)d_in[2];   // [T,B,1]
  const float* pstoch = (const float*)d_in[3];   // [B,S]
  const float* pdeter = (const float*)d_in[4];   // [B,D]
  const float* noise  = (const float*)d_in[5];   // [T,2,B,S]
  const float* W_emb  = (const float*)d_in[6];
  const float* b_emb  = (const float*)d_in[7];
  const float* W_ih   = (const float*)d_in[8];
  const float* b_ih   = (const float*)d_in[9];
  const float* W_hh   = (const float*)d_in[10];
  const float* b_hh   = (const float*)d_in[11];
  const float* Wp1    = (const float*)d_in[12];
  const float* bp1    = (const float*)d_in[13];
  const float* Wp2    = (const float*)d_in[14];
  const float* bp2    = (const float*)d_in[15];
  const float* Wq1    = (const float*)d_in[16];
  const float* bq1    = (const float*)d_in[17];
  const float* Wq2    = (const float*)d_in[18];
  const float* bq2    = (const float*)d_in[19];
  (void)in_sizes; (void)n_in; (void)out_size; (void)ws_size;

  // ---- workspace layout ----
  char* ws = (char*)d_ws;
  size_t off = 0;
  auto take = [&](size_t bytes) -> char* {
    char* p = ws + off;
    off = (off + bytes + 255) & ~(size_t)255;
    return p;
  };
  bf16_t* wemb_bf = (bf16_t*)take((size_t)DET * SA_DIM * 2);
  bf16_t* wih_bf  = (bf16_t*)take((size_t)3 * DET * DET * 2);
  bf16_t* whh_bf  = (bf16_t*)take((size_t)3 * DET * DET * 2);
  bf16_t* wp1_bf  = (bf16_t*)take((size_t)DET * DET * 2);
  bf16_t* wp2_bf  = (bf16_t*)take((size_t)2 * STO * DET * 2);
  bf16_t* wq1_bf  = (bf16_t*)take((size_t)DET * QK * 2);
  bf16_t* wq2_bf  = (bf16_t*)take((size_t)2 * STO * DET * 2);
  bf16_t* sa_bf   = (bf16_t*)take((size_t)BATCH * SA_DIM * 2);
  bf16_t* h_bf    = (bf16_t*)take((size_t)BATCH * DET * 2);
  float*  h_f32   = (float*) take((size_t)BATCH * DET * 4);
  bf16_t* x_bf    = (bf16_t*)take((size_t)BATCH * DET * 2);
  bf16_t* det_bf  = (bf16_t*)take((size_t)BATCH * DET * 2);
  bf16_t* ph_bf   = (bf16_t*)take((size_t)BATCH * DET * 2);
  bf16_t* qh_bf   = (bf16_t*)take((size_t)BATCH * DET * 2);
  float*  stoch_s = (float*) take((size_t)BATCH * STO * 4);
  float*  deter_s = (float*) take((size_t)BATCH * DET * 4);

  // ---- output layout (8 outputs concatenated, return order) ----
  const size_t SZ_S = (size_t)T_STEPS * BATCH * STO;
  const size_t SZ_D = (size_t)T_STEPS * BATCH * DET;
  float* outF      = (float*)d_out;
  float* o_pmean   = outF;
  float* o_pstd    = outF + SZ_S;
  float* o_pstoch  = outF + 2 * SZ_S;
  float* o_pdeter  = outF + 3 * SZ_S;
  float* o_qmean   = o_pdeter + SZ_D;
  float* o_qstd    = o_qmean + SZ_S;
  float* o_qstoch  = o_qstd + SZ_S;
  float* o_qdeter  = o_qstoch + SZ_S;

  const int BLK = 256;   // 8 waves/block (wave32)
  auto cvt = [&](const float* src, bf16_t* dst, size_t n) {
    cvt_f32_bf16<<<(unsigned)((n + BLK - 1) / BLK), BLK, 0, stream>>>(src, dst, (int)n);
  };

  // one-time per launch: weights fp32 -> bf16 (stay L2 resident, ~21 MB)
  cvt(W_emb, wemb_bf, (size_t)DET * SA_DIM);
  cvt(W_ih,  wih_bf,  (size_t)3 * DET * DET);
  cvt(W_hh,  whh_bf,  (size_t)3 * DET * DET);
  cvt(Wp1,   wp1_bf,  (size_t)DET * DET);
  cvt(Wp2,   wp2_bf,  (size_t)2 * STO * DET);
  cvt(Wq1,   wq1_bf,  (size_t)DET * QK);
  cvt(Wq2,   wq2_bf,  (size_t)2 * STO * DET);

  {
    int n = BATCH * STO + BATCH * DET;
    init_state<<<(n + BLK - 1) / BLK, BLK, 0, stream>>>(pstoch, pdeter, stoch_s, deter_s);
  }

  // grids: gemm blocks = (B/16) * (N/512); gru = (B/16)*8; head = B/16
  const int gemm_blocks = (BATCH / 16) * (DET / 512);     // 64
  const int gru_blocks  = (BATCH / 16) * 8;               // 256
  const int head_blocks = (BATCH / 16);                   // 32
  const int prep_n = BATCH * (SA_DIM + DET);

  const size_t smem_x  = (size_t)16 * SA_DIM * 2;         // 6 KB
  const size_t smem_d  = (size_t)16 * DET * 2;            // 32 KB
  const size_t smem_gru= 2 * smem_d;                      // 64 KB

  for (int t = 0; t < T_STEPS; ++t) {
    const float* obs_t = obs    + (size_t)t * BATCH * EMB;
    const float* act_t = action + (size_t)t * BATCH * ACTD;
    const float* nt_t  = nonterm+ (size_t)t * BATCH;
    const float* eps0  = noise  + ((size_t)t * 2 + 0) * BATCH * STO;
    const float* eps1  = noise  + ((size_t)t * 2 + 1) * BATCH * STO;
    const size_t oS = (size_t)t * BATCH * STO;
    const size_t oD = (size_t)t * BATCH * DET;

    prep_step<<<(prep_n + BLK - 1) / BLK, BLK, 0, stream>>>(
        stoch_s, act_t, nt_t, deter_s, sa_bf, h_bf, h_f32);

    // x = elu(sa @ W_emb^T + b_emb)
    gemm_bias_elu<<<gemm_blocks, BLK, smem_x, stream>>>(sa_bf, wemb_bf, b_emb,
                                                        x_bf, DET, SA_DIM);
    // deter = GRU(x, h)
    gru_step<<<gru_blocks, BLK, smem_gru, stream>>>(
        x_bf, h_bf, h_f32, wih_bf, whh_bf, b_ih, b_hh,
        deter_s, det_bf, o_pdeter + oD, o_qdeter + oD);

    // prior head
    gemm_bias_elu<<<gemm_blocks, BLK, smem_d, stream>>>(det_bf, wp1_bf, bp1,
                                                        ph_bf, DET, DET);
    head_step<<<head_blocks, BLK, smem_d, stream>>>(
        ph_bf, wp2_bf, bp2, eps0,
        o_pmean + oS, o_pstd + oS, o_pstoch + oS, nullptr);

    // posterior head (feeds next step's stoch)
    gemm_posterior_hidden<<<gemm_blocks, BLK, smem_d, stream>>>(det_bf, obs_t,
                                                                wq1_bf, bq1, qh_bf);
    head_step<<<head_blocks, BLK, smem_d, stream>>>(
        qh_bf, wq2_bf, bq2, eps1,
        o_qmean + oS, o_qstd + oS, o_qstoch + oS, stoch_s);
  }
}
